// LSTM_Decoder_47399259079351
// MI455X (gfx1250) — matrix operand order repaired and checked by reading, not compiled, not run on hardware
//
#include <hip/hip_runtime.h>
#include <hip/hip_bf16.h>

// ---------- types ----------
typedef __bf16 bf16;
typedef __attribute__((ext_vector_type(16))) __bf16 v16bf;
typedef __attribute__((ext_vector_type(8)))  __bf16 v8bf;
typedef __attribute__((ext_vector_type(8)))  float  v8f;

// ---------- problem constants ----------
constexpr int NB = 64;          // batch
constexpr int NT = 512;         // time steps
constexpr int NH = 512;         // hidden
constexpr int NG = 4 * NH;      // 2048 gate rows
constexpr int NK = 2 * NH;      // 1024 combined K (x | h)
constexpr int NBH = NB * NH;    // 32768

// ---------- helpers ----------
__device__ __forceinline__ bf16 f2bf(float f) {
  unsigned u = __builtin_bit_cast(unsigned, f);
  u += 0x7fffu + ((u >> 16) & 1u);           // round-to-nearest-even
  unsigned short s = (unsigned short)(u >> 16);
  return __builtin_bit_cast(bf16, s);
}
__device__ __forceinline__ float sigm(float x) { return 1.0f / (1.0f + __expf(-x)); }
__device__ __forceinline__ float tanh_f(float x) {
  float t = __expf(-2.0f * fabsf(x));
  float r = (1.0f - t) / (1.0f + t);
  return copysignf(r, x);
}

// ---------- prep kernels (fully parallel, run once per call) ----------
// Pack [Wih | Whh] per layer into row-major bf16 W[l][2048][1024]
__global__ void k_prep_w(const float* __restrict__ Wih, const float* __restrict__ Whh,
                         bf16* __restrict__ Wbf) {
  int idx = blockIdx.x * blockDim.x + threadIdx.x;      // 2*2048*1024 = 2^22 threads
  int l   = idx >> 21;
  int rem = idx & ((1 << 21) - 1);
  int n   = rem >> 10;
  int k   = rem & 1023;
  float v = (k < NH) ? Wih[(size_t)l * (NG * NH) + n * NH + k]
                     : Whh[(size_t)l * (NG * NH) + n * NH + (k - NH)];
  Wbf[idx] = f2bf(v);
}

__global__ void k_prep_bias(const float* __restrict__ bih, const float* __restrict__ bhh,
                            float* __restrict__ biasC) {
  int idx = blockIdx.x * blockDim.x + threadIdx.x;      // 2*2048
  biasC[idx] = bih[idx] + bhh[idx];
}

__global__ void k_prep_x(const float* __restrict__ x, bf16* __restrict__ Xbf) {
  int idx = blockIdx.x * blockDim.x + threadIdx.x;      // 64*512*512
  Xbf[idx] = f2bf(x[idx]);
}

// h0/c0 -> fp32 state (tail of d_out) and bf16 ping-pong parity-0 buffers
__global__ void k_prep_state(const float* __restrict__ h0, const float* __restrict__ c0,
                             float* __restrict__ hstate, float* __restrict__ cstate,
                             bf16* __restrict__ Hbf) {
  int idx = blockIdx.x * blockDim.x + threadIdx.x;      // 2*64*512 = 65536
  hstate[idx] = h0[idx];
  cstate[idx] = c0[idx];
  int l = idx >> 15;             // layer
  int w = idx & (NBH - 1);       // within [B*H]
  Hbf[(size_t)l * (2 * NBH) + w] = f2bf(h0[idx]);       // parity 0
}

// ---------- fragment loader for one 32-wide K chunk (explicit regs, no arrays)
__device__ __forceinline__ void load_chunk(
    int k0, const bf16* __restrict__ xin, int xstride, const bf16* __restrict__ hin,
    const bf16* __restrict__ W, int arow, int ncol, int half,
    v16bf& a, v16bf& b0, v16bf& b1, v16bf& b2, v16bf& b3)
{
  // A fragment: 16x32 bf16; lane holds 16 consecutive K [ab..ab+15]
  const int ab = k0 + half * 16;
  const bf16* asrc = (k0 < NH) ? (xin + (size_t)arow * xstride + ab)
                               : (hin + (size_t)arow * NH + (ab - NH));
  v8bf lo = *(const v8bf*)(asrc);
  v8bf hi = *(const v8bf*)(asrc + 8);
  #pragma unroll
  for (int i = 0; i < 8; ++i) { a[i] = lo[i]; a[i + 8] = hi[i]; }

  // B fragments: lane holds 16 consecutive K of W-row (512q + ncol)
  const int bk = k0 + half * 16;
  const bf16* wrow = W + (size_t)ncol * NK + bk;
  b0 = *(const v16bf*)(wrow + (size_t)(0 * NH) * NK);
  b1 = *(const v16bf*)(wrow + (size_t)(1 * NH) * NK);
  b2 = *(const v16bf*)(wrow + (size_t)(2 * NH) * NK);
  b3 = *(const v16bf*)(wrow + (size_t)(3 * NH) * NK);
}

#define WMMA_BF16(A, Bq, Cq) \
  __builtin_amdgcn_wmma_f32_16x16x32_bf16(false, (A), false, (Bq), (short)0, (Cq), false, false)

// ---------- fused per-step LSTM cell: GEMM (WMMA bf16) + gating ----------
// grid: 32 blocks (16-wide hidden column groups), block: 128 threads = 4 waves
// wave w owns batch rows [16w, 16w+16); computes 4 gate tiles (i,f,g,o) for its
// 16 hidden columns. Ping-pong (depth-1) software pipeline, 2 chunks/iteration.
__global__ void __launch_bounds__(128)
lstm_step(const bf16* __restrict__ xin, int xstride,   // A rows: batch stride in elems
          const bf16* __restrict__ hin,                // bf16 h (read parity)
          const bf16* __restrict__ W,                  // bf16 [2048][1024]
          const float* __restrict__ bias,              // fp32 [2048] (bih+bhh)
          float* __restrict__ cstate,                  // fp32 [B][H] (in/out)
          float* __restrict__ hstate,                  // fp32 [B][H] (out)
          float* __restrict__ out_seq,                 // fp32 d_out + t*H (or null)
          bf16* __restrict__ hbf_out,                  // bf16 h (write parity)
          bf16* __restrict__ xnext)                    // bf16 next-layer input (or null)
{
  const int lane = threadIdx.x & 31;
  const int half = lane >> 4;                 // 0: lanes 0-15, 1: lanes 16-31
  const int mrow = lane & 15;
  const int m0   = (threadIdx.x >> 5) << 4;   // batch tile base: 0,16,32,48
  const int n0   = blockIdx.x << 4;           // hidden column base (0..496)

  const int arow = m0 + mrow;                 // A-matrix row (batch index)
  const int ncol = n0 + mrow;                 // this lane's hidden column

  // ---- hoist gating-phase scalar loads: latency hides under the WMMA stream
  const float bi = bias[0 * NH + ncol];
  const float bf = bias[1 * NH + ncol];
  const float bg = bias[2 * NH + ncol];
  const float bo = bias[3 * NH + ncol];
  float cold[8];
  #pragma unroll
  for (int r = 0; r < 8; ++r)
    cold[r] = cstate[(size_t)(m0 + r + half * 8) * NH + ncol];

  v8f acc0, acc1, acc2, acc3;
  #pragma unroll
  for (int i = 0; i < 8; ++i) { acc0[i] = 0.0f; acc1[i] = 0.0f; acc2[i] = 0.0f; acc3[i] = 0.0f; }

  // ---- ping-pong pipelined K loop: two explicit fragment sets, no arrays
  v16bf aA, bA0, bA1, bA2, bA3;   // set A
  v16bf aB, bB0, bB1, bB2, bB3;   // set B
  load_chunk(0, xin, xstride, hin, W, arow, ncol, half, aA, bA0, bA1, bA2, bA3);

  #pragma unroll 1
  for (int k0 = 0; k0 < NK; k0 += 64) {
    // prefetch chunk k0+32 while computing chunk k0
    load_chunk(k0 + 32, xin, xstride, hin, W, arow, ncol, half,
               aB, bB0, bB1, bB2, bB3);
    acc0 = WMMA_BF16(aA, bA0, acc0);
    acc1 = WMMA_BF16(aA, bA1, acc1);
    acc2 = WMMA_BF16(aA, bA2, acc2);
    acc3 = WMMA_BF16(aA, bA3, acc3);
    // prefetch chunk k0+64 (wraps to 0 on last iter; dead value) while computing k0+32
    load_chunk((k0 + 64) & (NK - 1), xin, xstride, hin, W, arow, ncol, half,
               aA, bA0, bA1, bA2, bA3);
    acc0 = WMMA_BF16(aB, bB0, acc0);
    acc1 = WMMA_BF16(aB, bB1, acc1);
    acc2 = WMMA_BF16(aB, bB2, acc2);
    acc3 = WMMA_BF16(aB, bB3, acc3);
  }

  // ---- fused elementwise gating (C layout: VGPR r -> M = m0 + r + 8*half)
  #pragma unroll
  for (int r = 0; r < 8; ++r) {
    const int    br   = m0 + r + half * 8;
    const size_t sidx = (size_t)br * NH + ncol;
    float gi = acc0[r] + bi;
    float gf = acc1[r] + bf;
    float gg = acc2[r] + bg;
    float go = acc3[r] + bo;
    float cn = sigm(gf) * cold[r] + sigm(gi) * tanh_f(gg);
    float hn = sigm(go) * tanh_f(cn);
    cstate[sidx] = cn;
    hstate[sidx] = hn;
    bf16 hb = f2bf(hn);
    hbf_out[sidx] = hb;
    if (xnext)   xnext[sidx] = hb;
    if (out_seq) out_seq[(size_t)br * (NT * NH) + ncol] = hn;
  }
}

// ---------- host ----------
extern "C" void kernel_launch(void* const* d_in, const int* in_sizes, int n_in,
                              void* d_out, int out_size, void* d_ws, size_t ws_size,
                              hipStream_t stream) {
  const float* x   = (const float*)d_in[0];   // [64,512,512]
  const float* h0  = (const float*)d_in[1];   // [2,64,512]
  const float* c0  = (const float*)d_in[2];   // [2,64,512]
  const float* Wih = (const float*)d_in[3];   // [2,2048,512]
  const float* Whh = (const float*)d_in[4];   // [2,2048,512]
  const float* bih = (const float*)d_in[5];   // [2,2048]
  const float* bhh = (const float*)d_in[6];   // [2,2048]
  float* out = (float*)d_out;                 // out[64,512,512] | h_fin[2,64,512] | c_fin[2,64,512]

  // workspace layout (bytes)
  char* ws = (char*)d_ws;
  bf16*  Wbf   = (bf16*)(ws);                 //  8,388,608 B : [2][2048][1024] bf16
  bf16*  Xbf   = (bf16*)(ws + 8388608);       // 33,554,432 B : [64][512][512] bf16
  bf16*  Hbf   = (bf16*)(ws + 41943040);      //    262,144 B : [2 layer][2 parity][B*H] bf16
  bf16*  Xn    = (bf16*)(ws + 42205184);      //     65,536 B : [B*H] bf16 (layer0 -> layer1)
  float* biasC = (float*)(ws + 42270720);     //     16,384 B : [2][2048] fp32

  float* hstate = out + (size_t)NB * NT * NH; // h_fin region, live fp32 state
  float* cstate = hstate + 2 * NBH;           // c_fin region, live fp32 state

  // one-time (per call) parallel prep
  k_prep_w    <<<16384, 256, 0, stream>>>(Wih, Whh, Wbf);
  k_prep_bias <<<16,    256, 0, stream>>>(bih, bhh, biasC);
  k_prep_x    <<<65536, 256, 0, stream>>>(x, Xbf);
  k_prep_state<<<256,   256, 0, stream>>>(h0, c0, hstate, cstate, Hbf);

  // sequential recurrence: 2 fused WMMA kernels per time step
  for (int t = 0; t < NT; ++t) {
    const int p = t & 1;   // h ping-pong parity (read p, write 1-p)
    // layer 0: x from Xbf, writes Xn for layer 1
    lstm_step<<<32, 128, 0, stream>>>(
        Xbf + (size_t)t * NH, NT * NH,
        Hbf + (size_t)0 * 2 * NBH + (size_t)p * NBH,
        Wbf,
        biasC,
        cstate + 0 * NBH,
        hstate + 0 * NBH,
        nullptr,
        Hbf + (size_t)0 * 2 * NBH + (size_t)(1 - p) * NBH,
        Xn);
    // layer 1: x from Xn, writes sequence output
    lstm_step<<<32, 128, 0, stream>>>(
        Xn, NH,
        Hbf + (size_t)1 * 2 * NBH + (size_t)p * NBH,
        Wbf + (size_t)NG * NK,
        biasC + NG,
        cstate + 1 * NBH,
        hstate + 1 * NBH,
        out + (size_t)t * NH,
        Hbf + (size_t)1 * 2 * NBH + (size_t)(1 - p) * NBH,
        nullptr);
  }
}
